// ProductKeysMemory_36438502540076
// MI455X (gfx1250) — compile-verified, matrix operand order checked by reading, not run on hardware
//
#include <hip/hip_runtime.h>

// ---------------------------------------------------------------------------
// Types for CDNA5 WMMA (wave32): v_wmma_f32_16x16x32_bf16
// ---------------------------------------------------------------------------
typedef __bf16 bf16_t;
typedef __attribute__((ext_vector_type(16))) __bf16 v16bf;
typedef __attribute__((ext_vector_type(8)))  float  v8f;
typedef __attribute__((ext_vector_type(4)))  unsigned int vu4;

#define N_HEADS   4
#define QUERY_DIM 512
#define SUB_KEY_D 256
#define N_SUBKEYS 512
#define TOPK      16
#define SEQ_TOT   8192          // 4 * 2048
#define IN_DIM    512
#define QCOLS     2048          // N_HEADS * QUERY_DIM

static __device__ __forceinline__ v8f wmma_bf16(v16bf a, v16bf b, v8f c) {
    return __builtin_amdgcn_wmma_f32_16x16x32_bf16(
        /*neg_a=*/false, a, /*neg_b=*/false, b,
        /*c_mod=*/(short)0, c, /*reuse_a=*/false, /*reuse_b=*/false);
}

// Load a 16x32 bf16 A/B fragment for one lane.
// Layout (ISA 7.12.2, 16-bit A 16x32): lane L -> m = L&15, h = L>>4;
// elements 0..7  = K[h*8 .. h*8+7], elements 8..15 = K[16+h*8 .. 16+h*8+7].
// `base` must already point at (row m, k = kbase + h*8); both 16B runs contiguous.
static __device__ __forceinline__ v16bf load_frag(const bf16_t* base) {
    union { v16bf v; vu4 u[2]; } f;
    f.u[0] = *(const vu4*)(base);
    f.u[1] = *(const vu4*)(base + 16);
    return f.v;
}

// CDNA5 async global->LDS copy, 16B per lane, tracked by ASYNCcnt.
// the vdst VGPR holds the LDS byte address; low 32 bits of a generic LDS pointer are the
// allocation-relative offset (ISA 10.2 aperture rules).
static __device__ __forceinline__ void async_ld_b128(bf16_t* lds_dst, const bf16_t* gsrc) {
    unsigned loff = (unsigned)(size_t)lds_dst;
    asm volatile("global_load_async_to_lds_b128 %0, %1, off"
                 :: "v"(loff), "v"(gsrc) : "memory");
}
static __device__ __forceinline__ void wait_async_le1() {
    asm volatile("s_wait_asynccnt 0x1" ::: "memory");
}
static __device__ __forceinline__ void wait_async_0() {
    asm volatile("s_wait_asynccnt 0x0" ::: "memory");
}

// ---------------------------------------------------------------------------
// f32 -> bf16 conversion (one-shot prep of x, W_q, c1, c2)
// ---------------------------------------------------------------------------
__global__ __launch_bounds__(256) void cvt_bf16_kernel(const float* __restrict__ in,
                                                       bf16_t* __restrict__ out, int n) {
    int i = blockIdx.x * 256 + threadIdx.x;
    if (i < n) out[i] = (bf16_t)in[i];
}

// ---------------------------------------------------------------------------
// GEMM1: Q[8192,2048] = Xb[8192,512] * Wb[2048,512]^T + bias
// Block: 256 thr (8 waves). Block tile 128(M) x 64(N); wave w owns rows
// w*16..w*16+15. B tile (64x32 bf16) is staged in LDS once per block with
// double-buffered global_load_async_to_lds_b128 (ASYNCcnt). Grid (32,64).
// ---------------------------------------------------------------------------
#define BLD 40   // B tile row stride (bf16): 80B rows -> conflict-free ds_load_b128

__global__ __launch_bounds__(256) void gemm_q_kernel(const bf16_t* __restrict__ X,
                                                     const bf16_t* __restrict__ W,
                                                     const float*  __restrict__ bias,
                                                     float*        __restrict__ Q) {
    __shared__ bf16_t Bt[2][64 * BLD];

    const int tid  = threadIdx.x;
    const int w    = tid >> 5;
    const int lane = tid & 31;
    const int m    = lane & 15;
    const int h    = lane >> 4;
    const int m0   = blockIdx.y * 128 + w * 16;
    const int n0   = blockIdx.x * 64;

    // async staging map: thread t -> B row (t>>2), 16B chunk (t&3); 256 thr = 4KB tile
    const int srow = tid >> 2, schk = tid & 3;
    const bf16_t* gB = W + (size_t)(n0 + srow) * IN_DIM + schk * 8;
    bf16_t*       lB = &Bt[0][0] + srow * BLD + schk * 8;

    // prologue: stage k-step 0 into buffer 0
    async_ld_b128(lB, gB);

    const bf16_t* arow = X + (size_t)(m0 + m) * IN_DIM + h * 8;
    v8f acc[4];
#pragma unroll
    for (int nt = 0; nt < 4; ++nt) acc[nt] = (v8f)(0.0f);

    for (int ks = 0; ks < IN_DIM / 32; ++ks) {
        if (ks + 1 < IN_DIM / 32) {
            // stream next B tile into the other buffer while we compute
            async_ld_b128(lB + ((ks + 1) & 1) * 64 * BLD, gB + (ks + 1) * 32);
            wait_async_le1();        // current tile complete; next may be in flight
        } else {
            wait_async_0();
        }
        __syncthreads();             // all waves' async writes visible

        v16bf a = load_frag(arow + ks * 32);
        const bf16_t* bb = &Bt[ks & 1][0];
#pragma unroll
        for (int nt = 0; nt < 4; ++nt) {
            v16bf b = load_frag(bb + (nt * 16 + m) * BLD + h * 8);
            acc[nt] = wmma_bf16(a, b, acc[nt]);
        }
        __syncthreads();             // done reading before this buffer is overwritten
    }

    // D layout: lane -> n = lane&15; elem r -> m = h*8 + r
#pragma unroll
    for (int nt = 0; nt < 4; ++nt) {
        int n = n0 + nt * 16 + m;
        float bv = bias[n];
#pragma unroll
        for (int r = 0; r < 8; ++r) {
            int mm = m0 + h * 8 + r;
            Q[(size_t)mm * QCOLS + n] = acc[nt][r] + bv;
        }
    }
}

// ---------------------------------------------------------------------------
// BatchNorm stats: deterministic two-pass column reduction over 8192 rows.
// ---------------------------------------------------------------------------
__global__ __launch_bounds__(256) void bn_partial_kernel(const float* __restrict__ Q,
                                                         float* __restrict__ psum,
                                                         float* __restrict__ psumsq) {
    int col = blockIdx.x * 256 + threadIdx.x;
    int r0  = blockIdx.y * 128;
    float s = 0.f, s2 = 0.f;
    for (int r = 0; r < 128; ++r) {
        float v = Q[(size_t)(r0 + r) * QCOLS + col];
        s += v; s2 += v * v;
    }
    psum[(size_t)blockIdx.y * QCOLS + col]   = s;
    psumsq[(size_t)blockIdx.y * QCOLS + col] = s2;
}

__global__ __launch_bounds__(256) void bn_finalize_kernel(const float* __restrict__ psum,
                                                          const float* __restrict__ psumsq,
                                                          const float* __restrict__ gamma,
                                                          const float* __restrict__ beta,
                                                          float* __restrict__ scale,
                                                          float* __restrict__ shift) {
    int c = blockIdx.x * 256 + threadIdx.x;
    float s = 0.f, s2 = 0.f;
    for (int g = 0; g < 64; ++g) {
        s  += psum[(size_t)g * QCOLS + c];
        s2 += psumsq[(size_t)g * QCOLS + c];
    }
    const float inv_n = 1.0f / (float)SEQ_TOT;
    float mu  = s * inv_n;
    float var = s2 * inv_n - mu * mu;
    float sc  = gamma[c] * rsqrtf(var + 1e-5f);
    scale[c] = sc;
    shift[c] = beta[c] - mu * sc;
}

// ---------------------------------------------------------------------------
// Fused product-key kernel. One block = 16 rows, 256 threads (8 waves).
// Per head: BN-apply -> bf16 A tile in LDS; per side: WMMA scores (16x512)
// into LDS, wave-parallel iterative top-16; then combine, softmax, gather.
// ---------------------------------------------------------------------------
#define ROWS 16
#define ALD  520   // A tile row stride (bf16 elems), 1040B: 16B aligned, bank-spread
#define SLD  520   // score row stride (f32)
#define NEG_INF (-3.4e38f)

__global__ __launch_bounds__(256) void pkm_kernel(const float* __restrict__ Q,
                                                  const float* __restrict__ scale,
                                                  const float* __restrict__ shift,
                                                  const bf16_t* __restrict__ C1,
                                                  const bf16_t* __restrict__ C2,
                                                  const float* __restrict__ V,
                                                  float* __restrict__ Out) {
    __shared__ bf16_t Atile[ROWS * ALD];        // normalized q, one head (16x512)
    __shared__ float  Sl[ROWS * SLD];           // scores for one side (16x512)
    __shared__ float  topv[ROWS][2][TOPK];
    __shared__ int    topi[ROWS][2][TOPK];
    __shared__ float  selw[ROWS][TOPK];         // unnormalized softmax weights
    __shared__ int    selg[ROWS][TOPK];         // global value row indices

    const int tid  = threadIdx.x;
    const int w    = tid >> 5;
    const int lane = tid & 31;
    const int row0 = blockIdx.x * ROWS;
    const int m    = lane & 15;
    const int hh   = lane >> 4;

    // output accumulators: wave w owns rows 2w, 2w+1 (512 f32 each, as float4x4)
    float4 oacc[2][4];
#pragma unroll
    for (int a = 0; a < 2; ++a)
#pragma unroll
        for (int e = 0; e < 4; ++e) oacc[a][e] = make_float4(0.f, 0.f, 0.f, 0.f);

    for (int hd = 0; hd < N_HEADS; ++hd) {
        // ---- load + BN-normalize A tile (coalesced), convert to bf16 ----
#pragma unroll 4
        for (int i = 0; i < 32; ++i) {
            int e = tid + 256 * i;            // 0..8191
            int r = e >> 9, c = e & 511;
            int gc = hd * QUERY_DIM + c;
            float v = Q[(size_t)(row0 + r) * QCOLS + gc];
            Atile[r * ALD + c] = (bf16_t)(v * scale[gc] + shift[gc]);
        }
        __syncthreads();

        for (int side = 0; side < 2; ++side) {
            const bf16_t* C = side ? C2 : C1;

            // ---- WMMA scores: wave handles N-tiles w, w+8, w+16, w+24 ----
#pragma unroll
            for (int j = 0; j < 4; ++j) {
                int nt = w + 8 * j;
                v8f acc = (v8f)(0.0f);
                const bf16_t* crow = C + (size_t)(hd * N_SUBKEYS + nt * 16 + m) * SUB_KEY_D + hh * 8;
                const bf16_t* arow = Atile + m * ALD + side * SUB_KEY_D + hh * 8;
#pragma unroll
                for (int ks = 0; ks < SUB_KEY_D; ks += 32) {
                    v16bf a = load_frag(arow + ks);
                    v16bf b = load_frag(crow + ks);
                    acc = wmma_bf16(a, b, acc);
                }
#pragma unroll
                for (int r = 0; r < 8; ++r)
                    Sl[(hh * 8 + r) * SLD + nt * 16 + m] = acc[r];
            }
            __syncthreads();

            // ---- top-16 per row (wave handles rows w and w+8) ----
            for (int rr = 0; rr < 2; ++rr) {
                int r = w + 8 * rr;
                const float* srow = Sl + r * SLD;
                float v[16];
#pragma unroll
                for (int j2 = 0; j2 < 16; ++j2) v[j2] = srow[lane + 32 * j2];
                for (int t = 0; t < TOPK; ++t) {
                    float best = NEG_INF; int bidx = 0;
#pragma unroll
                    for (int j2 = 0; j2 < 16; ++j2)
                        if (v[j2] > best) { best = v[j2]; bidx = lane + 32 * j2; }
#pragma unroll
                    for (int off = 16; off > 0; off >>= 1) {
                        float ob = __shfl_xor(best, off, 32);
                        int   oi = __shfl_xor(bidx, off, 32);
                        if (ob > best || (ob == best && oi < bidx)) { best = ob; bidx = oi; }
                    }
                    if (lane == 0) { topv[r][side][t] = best; topi[r][side][t] = bidx; }
                    if ((bidx & 31) == lane) v[bidx >> 5] = NEG_INF;
                }
            }
            __syncthreads();
        }

        // ---- combine 16x16, top-16, softmax (wave handles rows 2w, 2w+1) ----
        float inv_sum[2];
        for (int rr = 0; rr < 2; ++rr) {
            int r = 2 * w + rr;
            float cs[8];
#pragma unroll
            for (int j2 = 0; j2 < 8; ++j2) {
                int p = lane + 32 * j2;
                cs[j2] = topv[r][0][p >> 4] + topv[r][1][p & 15];
            }
            float smax = 0.f, ssum = 0.f;
            for (int t = 0; t < TOPK; ++t) {
                float best = NEG_INF; int bidx = 0;
#pragma unroll
                for (int j2 = 0; j2 < 8; ++j2)
                    if (cs[j2] > best) { best = cs[j2]; bidx = lane + 32 * j2; }
#pragma unroll
                for (int off = 16; off > 0; off >>= 1) {
                    float ob = __shfl_xor(best, off, 32);
                    int   oi = __shfl_xor(bidx, off, 32);
                    if (ob > best || (ob == best && oi < bidx)) { best = ob; bidx = oi; }
                }
                if (t == 0) smax = best;
                float e = __expf(best - smax);
                ssum += e;
                if (lane == 0) {
                    int i1 = bidx >> 4, i2 = bidx & 15;
                    selw[r][t] = e;
                    selg[r][t] = topi[r][0][i1] * N_SUBKEYS + topi[r][1][i2];
                }
                if ((bidx & 31) == lane) cs[bidx >> 5] = NEG_INF;
            }
            inv_sum[rr] = 1.0f / ssum;
        }

        // ---- gather value rows, accumulate in registers (same wave as above) ----
        for (int rr = 0; rr < 2; ++rr) {
            int r = 2 * w + rr;
            for (int t = 0; t < TOPK; ++t) {
                float wt = selw[r][t] * inv_sum[rr];
                const float4* vp = (const float4*)(V + (size_t)selg[r][t] * IN_DIM);
                if (t + 1 < TOPK)   // hide the index->load dependency (global_prefetch_b8)
                    __builtin_prefetch(V + (size_t)selg[r][t + 1] * IN_DIM, 0, 1);
#pragma unroll
                for (int e = 0; e < 4; ++e) {
                    float4 x = vp[lane + 32 * e];
                    oacc[rr][e].x = fmaf(wt, x.x, oacc[rr][e].x);
                    oacc[rr][e].y = fmaf(wt, x.y, oacc[rr][e].y);
                    oacc[rr][e].z = fmaf(wt, x.z, oacc[rr][e].z);
                    oacc[rr][e].w = fmaf(wt, x.w, oacc[rr][e].w);
                }
            }
        }
        __syncthreads();   // before next head reuses LDS
    }

    // ---- write output (coalesced float4) ----
#pragma unroll
    for (int rr = 0; rr < 2; ++rr) {
        int r = row0 + 2 * w + rr;
        float4* op = (float4*)(Out + (size_t)r * IN_DIM);
#pragma unroll
        for (int e = 0; e < 4; ++e) op[lane + 32 * e] = oacc[rr][e];
    }
}

// ---------------------------------------------------------------------------
// Host launcher
// ---------------------------------------------------------------------------
extern "C" void kernel_launch(void* const* d_in, const int* in_sizes, int n_in,
                              void* d_out, int out_size, void* d_ws, size_t ws_size,
                              hipStream_t stream) {
    (void)in_sizes; (void)n_in; (void)out_size; (void)ws_size;
    const float* x      = (const float*)d_in[0];
    const float* W_q    = (const float*)d_in[1];
    const float* b_q    = (const float*)d_in[2];
    const float* gamma  = (const float*)d_in[3];
    const float* beta   = (const float*)d_in[4];
    const float* c1     = (const float*)d_in[5];
    const float* c2     = (const float*)d_in[6];
    const float* values = (const float*)d_in[7];
    float* out = (float*)d_out;

    // workspace carve-up
    char* p = (char*)d_ws;
    bf16_t* xb = (bf16_t*)p;  p += (size_t)SEQ_TOT * IN_DIM * sizeof(bf16_t);      // 8 MB
    bf16_t* wb = (bf16_t*)p;  p += (size_t)QCOLS * IN_DIM * sizeof(bf16_t);        // 2 MB
    bf16_t* c1b = (bf16_t*)p; p += (size_t)N_HEADS * N_SUBKEYS * SUB_KEY_D * sizeof(bf16_t);
    bf16_t* c2b = (bf16_t*)p; p += (size_t)N_HEADS * N_SUBKEYS * SUB_KEY_D * sizeof(bf16_t);
    float* q = (float*)p;     p += (size_t)SEQ_TOT * QCOLS * sizeof(float);        // 64 MB
    float* psum = (float*)p;  p += (size_t)64 * QCOLS * sizeof(float);
    float* psumsq = (float*)p;p += (size_t)64 * QCOLS * sizeof(float);
    float* scale = (float*)p; p += (size_t)QCOLS * sizeof(float);
    float* shift = (float*)p; p += (size_t)QCOLS * sizeof(float);

    // 1) bf16 conversions
    {
        int n = SEQ_TOT * IN_DIM;
        cvt_bf16_kernel<<<(n + 255) / 256, 256, 0, stream>>>(x, xb, n);
        n = QCOLS * IN_DIM;
        cvt_bf16_kernel<<<(n + 255) / 256, 256, 0, stream>>>(W_q, wb, n);
        n = N_HEADS * N_SUBKEYS * SUB_KEY_D;
        cvt_bf16_kernel<<<(n + 255) / 256, 256, 0, stream>>>(c1, c1b, n);
        cvt_bf16_kernel<<<(n + 255) / 256, 256, 0, stream>>>(c2, c2b, n);
    }

    // 2) q = x @ W_q^T + b_q  (bf16 WMMA, f32 accumulate, async B staging)
    {
        dim3 grid(QCOLS / 64, SEQ_TOT / 128);
        gemm_q_kernel<<<grid, 256, 0, stream>>>(xb, wb, b_q, q);
    }

    // 3) BatchNorm stats (deterministic two-pass)
    {
        dim3 grid(QCOLS / 256, SEQ_TOT / 128);
        bn_partial_kernel<<<grid, 256, 0, stream>>>(q, psum, psumsq);
        bn_finalize_kernel<<<QCOLS / 256, 256, 0, stream>>>(psum, psumsq, gamma, beta, scale, shift);
    }

    // 4) fused scores + top-k + softmax + gather
    {
        pkm_kernel<<<SEQ_TOT / ROWS, 256, 0, stream>>>(q, scale, shift, c1b, c2b, values, out);
    }
}
